// DecoderBlock_6330781794323
// MI455X (gfx1250) — matrix operand order repaired and checked by reading, not compiled, not run on hardware
//
#include <hip/hip_runtime.h>
#include <cstdint>
#include <cstddef>

// ---------------- problem constants (match reference) ----------------
#define D_MODEL_C  512
#define NHEAD_C    8
#define HEAD_DIM_C 64
#define D_STATE_C  16
#define D_INNER_C  1024
#define DT_RANK_C  32
#define D_CONV_C   4
#define BATCH_C    4
#define SEQ_C      1024
#define MEMLEN_C   1024
#define ROWS_C     (BATCH_C * SEQ_C)     // 4096 query rows

typedef __attribute__((ext_vector_type(16))) _Float16 v16h;
typedef __attribute__((ext_vector_type(8)))  _Float16 v8h;
typedef __attribute__((ext_vector_type(8)))  float    v8f;

// ---------------- f32 -> f16 conversion ----------------
__global__ void k_cvt_f16(const float* __restrict__ src, _Float16* __restrict__ dst, int n) {
  int i = blockIdx.x * 256 + threadIdx.x;
  if (i < n) dst[i] = (_Float16)src[i];
}

// ---------------- WMMA GEMM:  C[M,N] = act(scale * A[M,K] @ B[N,K]^T + bias[N]) ----------------
// A, B are f16 with K contiguous (lda/ldb in elements). ACT: 0=none 1=gelu(exact) 2=softplus.
// Each wave computes a 32x32 tile: 2 A frags x 2 B frags -> 4 WMMAs, each fragment reused twice
// (16 FLOP/byte of L2 traffic). M and N must be multiples of 32 (true for every call here).
template <int ACT, int OUT16>
__global__ void __launch_bounds__(128)
k_gemm(int M, int N, int K,
       const _Float16* __restrict__ A, int lda,
       const _Float16* __restrict__ B, int ldb,
       const float* __restrict__ bias, float scale,
       float* __restrict__ Cf, _Float16* __restrict__ Ch, int ldc)
{
  const int lane = threadIdx.x & 31;
  const int wave = threadIdx.x >> 5;
  const int ntn2 = N >> 5;                    // 32-wide column strips
  const int ntm2 = M >> 5;                    // 32-tall row strips
  const int tile = blockIdx.x * 4 + wave;
  if (tile >= ntm2 * ntn2) return;            // wave-uniform exit: EXEC stays all-ones for WMMA
  const int tm = (tile / ntn2) << 1;          // first 16-row tile
  const int tn = (tile - (tile / ntn2) * ntn2) << 1; // first 16-col tile
  const int hi = lane >> 4;                   // lane half-group (ISA 16x32 f16 fragment layout)
  const int lh = lane & 15;

  // A fragment: lane holds row; halves K in {k0+8*hi..+7} U {k0+16+8*hi..+7}
  const _Float16* ap0 = A + (size_t)(tm * 16 + lh) * lda + hi * 8;
  const _Float16* ap1 = ap0 + (size_t)16 * lda;
  // B fragment: lane holds column; halves K in {k0+8*hi..+7} U {k0+16+8*hi..+7}
  const _Float16* bp0 = B + (size_t)(tn * 16 + lh) * ldb + hi * 8;
  const _Float16* bp1 = bp0 + (size_t)16 * ldb;

  v8f acc00 = {0.f,0.f,0.f,0.f,0.f,0.f,0.f,0.f};
  v8f acc01 = acc00, acc10 = acc00, acc11 = acc00;
  for (int k0 = 0; k0 < K; k0 += 32) {
    v8h a0lo = *(const v8h*)(ap0 + k0);
    v8h a0hi = *(const v8h*)(ap0 + k0 + 16);
    v8h a1lo = *(const v8h*)(ap1 + k0);
    v8h a1hi = *(const v8h*)(ap1 + k0 + 16);
    v16h a0 = __builtin_shufflevector(a0lo, a0hi, 0,1,2,3,4,5,6,7,8,9,10,11,12,13,14,15);
    v16h a1 = __builtin_shufflevector(a1lo, a1hi, 0,1,2,3,4,5,6,7,8,9,10,11,12,13,14,15);
    v8h b0lo = *(const v8h*)(bp0 + k0);
    v8h b0hi = *(const v8h*)(bp0 + k0 + 16);
    v8h b1lo = *(const v8h*)(bp1 + k0);
    v8h b1hi = *(const v8h*)(bp1 + k0 + 16);
    v16h b0 = __builtin_shufflevector(b0lo, b0hi, 0,1,2,3,4,5,6,7,8,9,10,11,12,13,14,15);
    v16h b1 = __builtin_shufflevector(b1lo, b1hi, 0,1,2,3,4,5,6,7,8,9,10,11,12,13,14,15);
    acc00 = __builtin_amdgcn_wmma_f32_16x16x32_f16(false, a0, false, b0, (short)0, acc00, false, false);
    acc01 = __builtin_amdgcn_wmma_f32_16x16x32_f16(false, a0, false, b1, (short)0, acc01, false, false);
    acc10 = __builtin_amdgcn_wmma_f32_16x16x32_f16(false, a1, false, b0, (short)0, acc10, false, false);
    acc11 = __builtin_amdgcn_wmma_f32_16x16x32_f16(false, a1, false, b1, (short)0, acc11, false, false);
  }

  const int   mb0 = tm * 16 + hi * 8;         // D layout: VGPR v -> row v + 8*hi
  const int   mb1 = mb0 + 16;
  const int   n0 = tn * 16 + lh;
  const int   n1 = n0 + 16;
  const float bv0 = bias ? bias[n0] : 0.f;
  const float bv1 = bias ? bias[n1] : 0.f;
#pragma unroll
  for (int v = 0; v < 8; ++v) {
    float r00 = acc00[v] * scale + bv0;
    float r01 = acc01[v] * scale + bv1;
    float r10 = acc10[v] * scale + bv0;
    float r11 = acc11[v] * scale + bv1;
    if (ACT == 1) {            // exact GELU
      r00 = 0.5f * r00 * (1.f + erff(r00 * 0.7071067811865475f));
      r01 = 0.5f * r01 * (1.f + erff(r01 * 0.7071067811865475f));
      r10 = 0.5f * r10 * (1.f + erff(r10 * 0.7071067811865475f));
      r11 = 0.5f * r11 * (1.f + erff(r11 * 0.7071067811865475f));
    } else if (ACT == 2) {     // softplus
      r00 = (r00 > 20.f) ? r00 : log1pf(__expf(r00));
      r01 = (r01 > 20.f) ? r01 : log1pf(__expf(r01));
      r10 = (r10 > 20.f) ? r10 : log1pf(__expf(r10));
      r11 = (r11 > 20.f) ? r11 : log1pf(__expf(r11));
    }
    size_t o00 = (size_t)(mb0 + v) * ldc + n0;
    size_t o10 = (size_t)(mb1 + v) * ldc + n0;
    if (OUT16) {
      Ch[o00] = (_Float16)r00; Ch[o00 + 16] = (_Float16)r01;
      Ch[o10] = (_Float16)r10; Ch[o10 + 16] = (_Float16)r11;
    } else {
      Cf[o00] = r00; Cf[o00 + 16] = r01;
      Cf[o10] = r10; Cf[o10 + 16] = r11;
    }
  }
}

// ---------------- row softmax (f32 in -> f16 probs out) ----------------
__global__ void __launch_bounds__(256)
k_softmax(const float* __restrict__ S, _Float16* __restrict__ P, int cols) {
  const int row = blockIdx.x;
  const int t = threadIdx.x;
  const float* s = S + (size_t)row * cols;
  __shared__ float sh[256];
  float m = -1e30f;
  for (int c = t; c < cols; c += 256) m = fmaxf(m, s[c]);
  sh[t] = m; __syncthreads();
  for (int st = 128; st > 0; st >>= 1) { if (t < st) sh[t] = fmaxf(sh[t], sh[t + st]); __syncthreads(); }
  const float bm = sh[0]; __syncthreads();
  float sum = 0.f;
  for (int c = t; c < cols; c += 256) sum += __expf(s[c] - bm);
  sh[t] = sum; __syncthreads();
  for (int st = 128; st > 0; st >>= 1) { if (t < st) sh[t] += sh[t + st]; __syncthreads(); }
  const float inv = 1.f / sh[0];
  _Float16* p = P + (size_t)row * cols;
  for (int c = t; c < cols; c += 256) p[c] = (_Float16)(__expf(s[c] - bm) * inv);
}

// ---------------- transpose V: [B*L, 512] f32 -> [B, H, 64, L] f16 ----------------
__global__ void k_transpose_v(const float* __restrict__ v, _Float16* __restrict__ vt) {
  int i = blockIdx.x * 256 + threadIdx.x;
  if (i >= ROWS_C * D_MODEL_C) return;
  int col = i & (D_MODEL_C - 1);
  int row = i >> 9;                 // b*1024 + l
  int b = row >> 10, l = row & 1023;
  int h = col >> 6,  d = col & 63;
  vt[((((size_t)b * NHEAD_C + h) * HEAD_DIM_C + d) << 10) + l] = (_Float16)v[i];
}

// ---------------- residual + layernorm (cols = 512), optional f16 mirror ----------------
__global__ void __launch_bounds__(256)
k_add_ln(const float* __restrict__ x, const float* __restrict__ y,
         const float* __restrict__ g, const float* __restrict__ bta,
         float* __restrict__ out, _Float16* __restrict__ outh) {
  const int row = blockIdx.x;
  const int t = threadIdx.x;
  const float* xr = x + (size_t)row * D_MODEL_C;
  const float* yr = y + (size_t)row * D_MODEL_C;
  float a0 = xr[t] + yr[t];
  float a1 = xr[t + 256] + yr[t + 256];
  __shared__ float s1[256], s2[256];
  s1[t] = a0 + a1;
  s2[t] = a0 * a0 + a1 * a1;
  __syncthreads();
  for (int st = 128; st > 0; st >>= 1) {
    if (t < st) { s1[t] += s1[t + st]; s2[t] += s2[t + st]; }
    __syncthreads();
  }
  const float mu  = s1[0] * (1.f / D_MODEL_C);
  const float var = s2[0] * (1.f / D_MODEL_C) - mu * mu;
  const float inv = rsqrtf(var + 1e-5f);
  float r0 = (a0 - mu) * inv * g[t]       + bta[t];
  float r1 = (a1 - mu) * inv * g[t + 256] + bta[t + 256];
  out[(size_t)row * D_MODEL_C + t]       = r0;
  out[(size_t)row * D_MODEL_C + t + 256] = r1;
  if (outh) {
    outh[(size_t)row * D_MODEL_C + t]       = (_Float16)r0;
    outh[(size_t)row * D_MODEL_C + t + 256] = (_Float16)r1;
  }
}

// ---------------- depthwise causal conv (width 4) + bias + SiLU -> u (f16) ----------------
__global__ void k_conv_silu(const float* __restrict__ xz, const float* __restrict__ cw,
                            const float* __restrict__ cb, _Float16* __restrict__ uh) {
  int i = blockIdx.x * 256 + threadIdx.x;
  if (i >= ROWS_C * D_INNER_C) return;
  int c = i & (D_INNER_C - 1);
  int row = i >> 10;                // b*1024 + l
  int l = row & 1023, b = row >> 10;
  float acc = cb[c];
#pragma unroll
  for (int j = 0; j < D_CONV_C; ++j) {
    int ls = l - (D_CONV_C - 1) + j;
    if (ls >= 0) acc += cw[c * D_CONV_C + j] * xz[((size_t)(b * 1024 + ls)) * (2 * D_INNER_C) + c];
  }
  float s = acc / (1.f + __expf(-acc));
  uh[i] = (_Float16)s;
}

// ---------------- selective scan: state[16] in registers, Bm/Cm staged in LDS ----------------
__global__ void __launch_bounds__(256)
k_scan(const _Float16* __restrict__ uh, const _Float16* __restrict__ dth,
       const _Float16* __restrict__ xdblh, const float* __restrict__ A_log,
       const float* __restrict__ D_p, const float* __restrict__ xz,
       _Float16* __restrict__ yh) {
  const int b = blockIdx.x;
  const int d = blockIdx.y * 256 + threadIdx.x;
  float A[D_STATE_C];
#pragma unroll
  for (int s = 0; s < D_STATE_C; ++s) A[s] = -__expf(A_log[d * D_STATE_C + s]);
  const float Dp = D_p[d];
  float h[D_STATE_C];
#pragma unroll
  for (int s = 0; s < D_STATE_C; ++s) h[s] = 0.f;
  __shared__ float BC[2 * D_STATE_C];
  for (int l = 0; l < SEQ_C; ++l) {
    const size_t row = (size_t)b * SEQ_C + l;
    if (threadIdx.x < 32) BC[threadIdx.x] = (float)xdblh[row * 64 + DT_RANK_C + threadIdx.x];
    __syncthreads();
    const float dt = (float)dth[row * D_INNER_C + d];
    const float uv = (float)uh[row * D_INNER_C + d];
    float y = 0.f;
#pragma unroll
    for (int s = 0; s < D_STATE_C; ++s) {
      float dA = __expf(dt * A[s]);
      h[s] = h[s] * dA + dt * BC[s] * uv;
      y += h[s] * BC[D_STATE_C + s];
    }
    y += uv * Dp;
    const float z = xz[row * (2 * D_INNER_C) + D_INNER_C + d];
    y *= z / (1.f + __expf(-z));
    yh[row * D_INNER_C + d] = (_Float16)y;
    __syncthreads();
  }
}

// ---------------- host orchestration ----------------
extern "C" void kernel_launch(void* const* d_in, const int* in_sizes, int n_in,
                              void* d_out, int out_size, void* d_ws, size_t ws_size,
                              hipStream_t stream) {
  (void)in_sizes; (void)n_in; (void)out_size;
  const float* x      = (const float*)d_in[0];
  const float* memory = (const float*)d_in[1];
  const float* Wq = (const float*)d_in[2];   const float* bq = (const float*)d_in[3];
  const float* Wk = (const float*)d_in[4];   const float* bk = (const float*)d_in[5];
  const float* Wv = (const float*)d_in[6];   const float* bv = (const float*)d_in[7];
  const float* Wo = (const float*)d_in[8];   const float* bo = (const float*)d_in[9];
  const float* ln1_g = (const float*)d_in[10]; const float* ln1_b = (const float*)d_in[11];
  const float* ln2_g = (const float*)d_in[12]; const float* ln2_b = (const float*)d_in[13];
  const float* ln3_g = (const float*)d_in[14]; const float* ln3_b = (const float*)d_in[15];
  const float* W_in   = (const float*)d_in[16];
  const float* conv_w = (const float*)d_in[17]; const float* conv_b = (const float*)d_in[18];
  const float* W_xproj = (const float*)d_in[19];
  const float* W_dt = (const float*)d_in[20];  const float* b_dt = (const float*)d_in[21];
  const float* A_log = (const float*)d_in[22]; const float* D_p  = (const float*)d_in[23];
  const float* W_out = (const float*)d_in[24];
  const float* W1 = (const float*)d_in[25];    const float* b1 = (const float*)d_in[26];
  const float* W2 = (const float*)d_in[27];    const float* b2 = (const float*)d_in[28];

  // ---- workspace arena ----
  char* p = (char*)d_ws;
  auto alloc = [&](size_t bytes) -> char* {
    char* r = p; p += (bytes + 255) & ~(size_t)255; return r;
  };
  const size_t NM = (size_t)ROWS_C * D_MODEL_C;      // 2M elements
  const size_t NI = (size_t)ROWS_C * D_INNER_C;      // 4M elements

  _Float16* xh    = (_Float16*)alloc(NM * 2);
  _Float16* memh  = (_Float16*)alloc(NM * 2);
  _Float16* Wqh   = (_Float16*)alloc((size_t)512 * 512 * 2);
  _Float16* Wkh   = (_Float16*)alloc((size_t)512 * 512 * 2);
  _Float16* Wvh   = (_Float16*)alloc((size_t)512 * 512 * 2);
  _Float16* Woh   = (_Float16*)alloc((size_t)512 * 512 * 2);
  _Float16* Winh  = (_Float16*)alloc((size_t)2048 * 512 * 2);
  _Float16* Wxph  = (_Float16*)alloc((size_t)64 * 1024 * 2);
  _Float16* Wdth  = (_Float16*)alloc((size_t)1024 * 32 * 2);
  _Float16* Wouth = (_Float16*)alloc((size_t)512 * 1024 * 2);
  _Float16* W1h   = (_Float16*)alloc((size_t)2048 * 512 * 2);
  _Float16* W2h   = (_Float16*)alloc((size_t)512 * 2048 * 2);
  _Float16* qh    = (_Float16*)alloc(NM * 2);
  _Float16* kh    = (_Float16*)alloc(NM * 2);
  float*    tmpf  = (float*)   alloc(NM * 4);        // v_f32 -> attnproj -> mamba_out (reused)
  _Float16* vth   = (_Float16*)alloc(NM * 2);        // [B,H,64,L]
  float*    sc    = (float*)   alloc((size_t)SEQ_C * MEMLEN_C * 4);
  _Float16* pr    = (_Float16*)alloc((size_t)SEQ_C * MEMLEN_C * 2);
  _Float16* attnh = (_Float16*)alloc(NM * 2);
  float*    x1    = (float*)   alloc(NM * 4);
  _Float16* x1h   = (_Float16*)alloc(NM * 2);
  float*    xz    = (float*)   alloc(NI * 2 * 4);    // [4096, 2048]; reused for ffn
  _Float16* uh    = (_Float16*)alloc(NI * 2);
  _Float16* xdblh = (_Float16*)alloc((size_t)ROWS_C * 64 * 2);
  _Float16* dth   = (_Float16*)alloc(NI * 2);
  _Float16* yh    = (_Float16*)alloc(NI * 2);
  float*    x2    = (float*)   alloc(NM * 4);
  _Float16* x2h   = (_Float16*)alloc(NM * 2);
  _Float16* h1h   = (_Float16*)alloc(NI * 2 * 2);    // [4096, 2048] f16
  float*    ffn   = xz;                              // alias: xz dead after the scan
  if ((size_t)(p - (char*)d_ws) > ws_size) return;   // workspace too small: bail

  auto cvt = [&](const float* s, _Float16* d, int n) {
    k_cvt_f16<<<(n + 255) / 256, 256, 0, stream>>>(s, d, n);
  };
  auto gblocks = [](int M, int N) { return ((M >> 5) * (N >> 5) + 3) / 4; };

  // ---- precision staging ----
  cvt(x, xh, (int)NM);            cvt(memory, memh, (int)NM);
  cvt(Wq, Wqh, 512 * 512);        cvt(Wk, Wkh, 512 * 512);
  cvt(Wv, Wvh, 512 * 512);        cvt(Wo, Woh, 512 * 512);
  cvt(W_in, Winh, 2048 * 512);    cvt(W_xproj, Wxph, 64 * 1024);
  cvt(W_dt, Wdth, 1024 * 32);     cvt(W_out, Wouth, 512 * 1024);
  cvt(W1, W1h, 2048 * 512);       cvt(W2, W2h, 512 * 2048);

  // ---- cross-attention ----
  k_gemm<0,1><<<gblocks(ROWS_C,512),128,0,stream>>>(ROWS_C,512,512, xh,512,  Wqh,512, bq,1.f, nullptr, qh, 512);
  k_gemm<0,1><<<gblocks(ROWS_C,512),128,0,stream>>>(ROWS_C,512,512, memh,512,Wkh,512, bk,1.f, nullptr, kh, 512);
  k_gemm<0,0><<<gblocks(ROWS_C,512),128,0,stream>>>(ROWS_C,512,512, memh,512,Wvh,512, bv,1.f, tmpf, nullptr, 512);
  k_transpose_v<<<(int)((NM + 255) / 256), 256, 0, stream>>>(tmpf, vth);
  for (int b = 0; b < BATCH_C; ++b)
    for (int h = 0; h < NHEAD_C; ++h) {
      const _Float16* qp = qh + (size_t)b * SEQ_C * D_MODEL_C + h * HEAD_DIM_C;
      const _Float16* kp = kh + (size_t)b * MEMLEN_C * D_MODEL_C + h * HEAD_DIM_C;
      const _Float16* vp = vth + ((size_t)(b * NHEAD_C + h) * HEAD_DIM_C) * MEMLEN_C;
      _Float16* op = attnh + (size_t)b * SEQ_C * D_MODEL_C + h * HEAD_DIM_C;
      k_gemm<0,0><<<gblocks(SEQ_C,MEMLEN_C),128,0,stream>>>(SEQ_C,MEMLEN_C,HEAD_DIM_C, qp,512, kp,512,
                                                            nullptr, 0.125f, sc, nullptr, MEMLEN_C);
      k_softmax<<<SEQ_C,256,0,stream>>>(sc, pr, MEMLEN_C);
      k_gemm<0,1><<<gblocks(SEQ_C,HEAD_DIM_C),128,0,stream>>>(SEQ_C,HEAD_DIM_C,MEMLEN_C, pr,MEMLEN_C,
                                                              vp,MEMLEN_C, nullptr,1.f, nullptr, op, 512);
    }
  k_gemm<0,0><<<gblocks(ROWS_C,512),128,0,stream>>>(ROWS_C,512,512, attnh,512, Woh,512, bo,1.f, tmpf, nullptr, 512);
  k_add_ln<<<ROWS_C,256,0,stream>>>(x, tmpf, ln1_g, ln1_b, x1, x1h);

  // ---- mamba ----
  k_gemm<0,0><<<gblocks(ROWS_C,2048),128,0,stream>>>(ROWS_C,2048,512, x1h,512, Winh,512, nullptr,1.f, xz, nullptr, 2048);
  k_conv_silu<<<(int)((NI + 255) / 256), 256, 0, stream>>>(xz, conv_w, conv_b, uh);
  k_gemm<0,1><<<gblocks(ROWS_C,64),128,0,stream>>>(ROWS_C,64,1024, uh,1024, Wxph,1024, nullptr,1.f, nullptr, xdblh, 64);
  k_gemm<2,1><<<gblocks(ROWS_C,1024),128,0,stream>>>(ROWS_C,1024,32, xdblh,64, Wdth,32, b_dt,1.f, nullptr, dth, 1024);
  k_scan<<<dim3(BATCH_C, D_INNER_C / 256), 256, 0, stream>>>(uh, dth, xdblh, A_log, D_p, xz, yh);
  k_gemm<0,0><<<gblocks(ROWS_C,512),128,0,stream>>>(ROWS_C,512,1024, yh,1024, Wouth,1024, nullptr,1.f, tmpf, nullptr, 512);
  k_add_ln<<<ROWS_C,256,0,stream>>>(x1, tmpf, ln2_g, ln2_b, x2, x2h);

  // ---- ffn ----
  k_gemm<1,1><<<gblocks(ROWS_C,2048),128,0,stream>>>(ROWS_C,2048,512, x2h,512, W1h,512, b1,1.f, nullptr, h1h, 2048);
  k_gemm<0,0><<<gblocks(ROWS_C,512),128,0,stream>>>(ROWS_C,512,2048, h1h,2048, W2h,2048, b2,1.f, ffn, nullptr, 512);
  k_add_ln<<<ROWS_C,256,0,stream>>>(x2, ffn, ln3_g, ln3_b, (float*)d_out, nullptr);
}
